// StratifiedDQE_6622839570795
// MI455X (gfx1250) — matrix operand order repaired and checked by reading, not compiled, not run on hardware
//
#include <hip/hip_runtime.h>
#include <hip/hip_bf16.h>

#define NN   20000
#define EE   320000
#define INF  7
#define HH   256
#define DD   128
#define KK   4
#define NI   8
#define NP   8
#define TT   4
#define NLAY 4
#define EPSV 1e-6f

typedef __attribute__((ext_vector_type(16))) _Float16     v16h;
typedef __attribute__((ext_vector_type(8)))  float        v8f;
typedef __attribute__((ext_vector_type(4)))  float        vf4;
typedef __attribute__((ext_vector_type(4)))  unsigned int u32x4;
typedef __attribute__((ext_vector_type(8)))  int          i32x8;
typedef __attribute__((ext_vector_type(4)))  int          i32x4;

// ---------------- small helpers ----------------

__device__ __forceinline__ float wave_sum32(float v) {
#pragma unroll
  for (int m = 16; m >= 1; m >>= 1) v += __shfl_xor(v, m, 32);
  return v;
}

__device__ __forceinline__ float tan_k(float x, float k) {
  float s  = sqrtf(fabsf(k) + 1e-15f);
  float sx = s * x;
  if (k > 0.f) return tanf(fminf(fmaxf(sx, -1.55f), 1.55f)) / s;
  if (k < 0.f) return tanhf(sx) / s;
  return x;
}

__device__ __forceinline__ float arctan_k(float x, float k) {
  float s  = sqrtf(fabsf(k) + 1e-15f);
  float sx = s * x;
  if (k > 0.f) return atanf(sx) / s;
  if (k < 0.f) return atanhf(fminf(fmaxf(sx, -1.f + 1e-6f), 1.f - 1e-6f)) / s;
  return x;
}

__device__ __forceinline__ float sigm(float x) { return 1.f / (1.f + expf(-x)); }

// ---------------- utility kernels ----------------

__global__ void zero_f32(float* p, int n) {
  int i = blockIdx.x * blockDim.x + threadIdx.x;
  if (i < n) p[i] = 0.f;
}

__global__ void copy_f32(const float* a, float* b, int n) {
  int i = blockIdx.x * blockDim.x + threadIdx.x;
  if (i < n) b[i] = a[i];
}

__global__ void deg_kernel(const int* ei, float* deg) {
  int e = blockIdx.x * blockDim.x + threadIdx.x;
  if (e < EE) unsafeAtomicAdd(&deg[ei[EE + e]], 1.f);
}

// Wcat (oc x 512) f16 = [lin_l_w | lin_r_w]
__global__ void wcat_prep(const float* wl, const float* wr, _Float16* wcat, int oc) {
  int i = blockIdx.x * blockDim.x + threadIdx.x;
  if (i >= oc * 512) return;
  int o = i >> 9, k = i & 511;
  float v = (k < 256) ? wl[o * 256 + k] : wr[o * 256 + (k - 256)];
  wcat[i] = (_Float16)v;
}

// h0 = x @ Win^T + b ; also f16 copy into Acat[:,256:512)
__global__ void input_proj(const float* x, const float* w, const float* b,
                           float* h, _Float16* acat) {
  int i = blockIdx.x * blockDim.x + threadIdx.x;
  if (i >= NN * HH) return;
  int n = i >> 8, o = i & 255;
  float acc = b[o];
#pragma unroll
  for (int j = 0; j < INF; ++j) acc += x[n * INF + j] * w[o * INF + j];
  h[i] = acc;
  acat[(size_t)n * 512 + 256 + o] = (_Float16)acc;
}

// agg[dst] += h[src]  (4 floats per thread)
__global__ void scatter_add(const float* __restrict__ h, const int* __restrict__ ei,
                            float* __restrict__ agg) {
  long long idx = (long long)blockIdx.x * blockDim.x + threadIdx.x;
  if (idx >= (long long)EE * 64) return;
  int e = (int)(idx >> 6);
  int c = ((int)idx & 63) << 2;
  int src = ei[e], dst = ei[EE + e];
  vf4 v = *(const vf4*)(h + (size_t)src * HH + c);
  float* p = agg + (size_t)dst * HH + c;
  unsafeAtomicAdd(p + 0, v.x);
  unsafeAtomicAdd(p + 1, v.y);
  unsafeAtomicAdd(p + 2, v.z);
  unsafeAtomicAdd(p + 3, v.w);
}

// Acat[:,0:256) = f16(agg / max(deg,1))
__global__ void agg_norm(const float* agg, const float* deg, _Float16* acat) {
  int i = blockIdx.x * blockDim.x + threadIdx.x;
  if (i >= NN * HH) return;
  int n = i >> 8, o = i & 255;
  float d = fmaxf(deg[n], 1.f);
  acat[(size_t)n * 512 + o] = (_Float16)(agg[i] / d);
}

// ---------------- WMMA GEMM: C(NN x oc) = Acat(NN x 512,f16) @ Wcat^T(oc x 512,f16) ----------------
// Block: 8 waves, one 64-col weight supertile staged in LDS via the Tensor Data Mover.
// Each wave computes a 16(M) x 64(N) tile: 4 accumulators, K-loop of 16 wmma ops,
// A fragments from global, B fragments from LDS (ds_load_b128).
__global__ __launch_bounds__(256) void gemm_wmma(const _Float16* __restrict__ A,
                                                 const _Float16* __restrict__ W,
                                                 float* __restrict__ C, int oc) {
  __shared__ _Float16 sW[64 * 512];                  // 64 KB weight supertile
  const int lane = threadIdx.x & 31;
  const int wave = threadIdx.x >> 5;
  const int ntn  = oc >> 6;                          // 64-col supertiles
  const int bm   = blockIdx.x / ntn;
  const int n0   = (blockIdx.x % ntn) << 6;

  // ---- stage W rows [n0, n0+64) (contiguous 64 KB) into LDS via TDM ----
#if __has_builtin(__builtin_amdgcn_tensor_load_to_lds) && __has_builtin(__builtin_amdgcn_s_wait_tensorcnt)
  if (threadIdx.x < 32) {                            // one wave issues the descriptor
    unsigned lds_off = (unsigned)(size_t)(&sW[0]);   // LDS byte offset (low 32 bits of aperture addr)
    unsigned long long ga = (unsigned long long)(const void*)(W + (size_t)n0 * 512);
    u32x4 g0;
    g0[0] = 1u;                                      // count=1 (valid user descriptor)
    g0[1] = lds_off;                                 // lds_addr
    g0[2] = (unsigned)(ga & 0xffffffffu);            // global_addr[31:0]
    g0[3] = (unsigned)((ga >> 32) & 0x01ffffffu) | 0x80000000u;  // addr[56:32] | type=2
    i32x8 g1;
    g1[0] = 0x00020000;                              // data_size=4B, workgroup_mask=0
    g1[1] = (int)(16384u << 16);                     // tensor_dim0 = 16384 dwords (lo16)
    g1[2] = (int)(1u << 16);                         // tensor_dim0 hi | tensor_dim1 = 1
    g1[3] = (int)(16384u << 16);                     // tensor_dim1 hi | tile_dim0 = 16384
    g1[4] = 1;                                       // tile_dim1 = 1, tile_dim2 = 0
    g1[5] = 16384;                                   // tensor_dim0_stride lo32
    g1[6] = 0;                                       // stride0 hi | tensor_dim1_stride lo16
    g1[7] = 0;
    i32x4 g2; g2[0] = 1; g2[1] = 0; g2[2] = 0; g2[3] = 0;   // tensor_dim2 = 1 (unused)
    i32x4 g3; g3[0] = 0; g3[1] = 0; g3[2] = 0; g3[3] = 0;
#if defined(__clang_major__) && (__clang_major__ >= 23)
    i32x8 gz = {};
    __builtin_amdgcn_tensor_load_to_lds(g0, g1, g2, g3, gz, 0);
#else
    __builtin_amdgcn_tensor_load_to_lds(g0, g1, g2, g3, 0);
#endif
    __builtin_amdgcn_s_wait_tensorcnt((short)0);
  }
#else
  // fallback: cooperative vector copy
  for (int i = threadIdx.x; i < 64 * 512 / 8; i += 256)
    ((vf4*)sW)[i] = ((const vf4*)(W + (size_t)n0 * 512))[i];
#endif
  __syncthreads();

  const int mt = bm * 8 + wave;
  if (mt < (NN >> 4)) {
    const int m0  = mt << 4;
    const int row = m0 + (lane & 15);
    const int kA  = (lane >> 4) << 3;                // 0 or 8   (A: 16-bit 16x32 layout)
    const int kB  = (lane >> 4) << 3;                // 0 or 8   (B: 16-bit 32x16 layout)
    v8f acc[4] = {v8f{}, v8f{}, v8f{}, v8f{}};
    for (int k0 = 0; k0 < 512; k0 += 32) {
      v16h a;
      const _Float16* ap = A + (size_t)row * 512 + k0 + kA;
      ((vf4*)&a)[0] = *(const vf4*)(ap);
      ((vf4*)&a)[1] = *(const vf4*)(ap + 16);
#pragma unroll
      for (int s = 0; s < 4; ++s) {
        const _Float16* wp = &sW[(size_t)(s * 16 + (lane & 15)) * 512 + k0 + kB];
        v16h b;
        ((vf4*)&b)[0] = ((const vf4*)wp)[0];
        ((vf4*)&b)[1] = ((const vf4*)wp)[2];
        acc[s] = __builtin_amdgcn_wmma_f32_16x16x32_f16(
            false, a, false, b, (short)0, acc[s], false, false);
      }
    }
#pragma unroll
    for (int s = 0; s < 4; ++s) {
#pragma unroll
      for (int j = 0; j < 8; ++j) {
        int m = m0 + j + ((lane >> 4) << 3);
        int n = n0 + s * 16 + (lane & 15);
        C[(size_t)m * oc + n] = acc[s][j];
      }
    }
  }
}

// ---------------- bias + SiLU + LayerNorm epilogue (block per node) ----------------
__global__ void epilogue(const float* __restrict__ h2, const float* bl, const float* br,
                         const float* g, const float* be,
                         float* __restrict__ hout, _Float16* __restrict__ acat,
                         float* __restrict__ emb, int oc, int last) {
  __shared__ float red[256];
  int n = blockIdx.x, o = threadIdx.x;
  float v = h2[(size_t)n * oc + o] + bl[o] + br[o];
  v = v * sigm(v);                                    // silu
  red[o] = v; __syncthreads();
  for (int st = oc >> 1; st > 0; st >>= 1) { if (o < st) red[o] += red[o + st]; __syncthreads(); }
  float mu = red[0] / oc; __syncthreads();
  float dv = v - mu;
  red[o] = dv * dv; __syncthreads();
  for (int st = oc >> 1; st > 0; st >>= 1) { if (o < st) red[o] += red[o + st]; __syncthreads(); }
  float var = red[0] / oc;
  float hn = dv * rsqrtf(var + 1e-5f) * g[o] + be[o];
  if (last) {
    emb[(size_t)n * DD + o] = hn;
  } else {
    hout[(size_t)n * oc + o] = hn;
    acat[(size_t)n * 512 + 256 + o] = (_Float16)hn;
  }
}

// ---------------- head prep: pm = clip(exp_map(protos)), w_hat, kappas ----------------
__global__ void head_prep(const float* kraw, const float* pw, const float* protos,
                          float* w_hat, float* pm, float* pm_y2, float* out_kappas) {
  __shared__ float red[128];
  int d = threadIdx.x;
  if (d < KK) out_kappas[d] = 2.f * tanhf(kraw[d]);
  for (int k = 0; k < KK; ++k) {
    float kap = 2.f * tanhf(kraw[k]);
    float s   = sqrtf(fabsf(kap) + 1e-15f);
    float maxn = (kap < 0.f) ? (1.f - 1e-5f) / s : 1e15f;
    // normalized projection vector
    float wv = pw[k * DD + d];
    red[d] = wv * wv; __syncthreads();
    for (int st = 64; st > 0; st >>= 1) { if (d < st) red[d] += red[d + st]; __syncthreads(); }
    float wn = fmaxf(sqrtf(red[0]), EPSV); __syncthreads();
    w_hat[k * DD + d] = wv / wn;
    for (int p = 0; p < NP; ++p) {
      float tv = protos[(size_t)(k * NP + p) * DD + d];
      red[d] = tv * tv; __syncthreads();
      for (int st = 64; st > 0; st >>= 1) { if (d < st) red[d] += red[d + st]; __syncthreads(); }
      float nrm = fmaxf(sqrtf(red[0]), EPSV); __syncthreads();
      float tk    = tan_k(nrm, kap);
      float pnorm = fabsf(tk);
      float scale = fminf(1.f, maxn / fmaxf(pnorm, EPSV));
      float pmv   = tv * (tk / nrm) * scale;
      pm[(size_t)(k * NP + p) * DD + d] = pmv;
      red[d] = pmv * pmv; __syncthreads();
      for (int st = 64; st > 0; st >>= 1) { if (d < st) red[d] += red[d + st]; __syncthreads(); }
      if (d == 0) pm_y2[k * NP + p] = red[0];
      __syncthreads();
    }
  }
}

// ---------------- head: one wave32 per node ----------------
__global__ __launch_bounds__(256) void head_kernel(
    const float* __restrict__ emb, const float* __restrict__ centroids,
    const float* __restrict__ log_temp, const float* __restrict__ kraw,
    const float* __restrict__ icen, const float* __restrict__ lhw,
    const float* __restrict__ lit, const float* __restrict__ w_hat,
    const float* __restrict__ pm, const float* __restrict__ pm_y2,
    const float* __restrict__ lpt, const float* __restrict__ hw,
    const float* __restrict__ hb, float* __restrict__ out_pred,
    float* __restrict__ out_sa) {
  const int lane = threadIdx.x & 31;
  const int node = (int)((blockIdx.x * (size_t)blockDim.x + threadIdx.x) >> 5);
  if (node >= NN) return;

  float e[4];
#pragma unroll
  for (int j = 0; j < 4; ++j) e[j] = emb[(size_t)node * DD + lane + 32 * j];
  float e2 = 0.f;
#pragma unroll
  for (int j = 0; j < 4; ++j) e2 += e[j] * e[j];
  e2 = wave_sum32(e2);
  float en = fmaxf(sqrtf(e2), EPSV);

  // soft assignment over centroids
  float temp = fminf(fmaxf(expf(log_temp[0]), 0.01f), 100.f);
  float sq[KK];
#pragma unroll
  for (int k = 0; k < KK; ++k) {
    float d2 = 0.f;
#pragma unroll
    for (int j = 0; j < 4; ++j) {
      float df = e[j] - centroids[k * DD + lane + 32 * j];
      d2 += df * df;
    }
    sq[k] = wave_sum32(d2);
  }
  float mx = -temp * sq[0];
#pragma unroll
  for (int k = 1; k < KK; ++k) mx = fmaxf(mx, -temp * sq[k]);
  float sa[KK], ssum = 0.f;
#pragma unroll
  for (int k = 0; k < KK; ++k) { sa[k] = expf(-temp * sq[k] - mx); ssum += sa[k]; }
#pragma unroll
  for (int k = 0; k < KK; ++k) sa[k] /= ssum;

  float pred[TT] = {0.f, 0.f, 0.f, 0.f};

#pragma unroll 1
  for (int k = 0; k < KK; ++k) {
    float kap = 2.f * tanhf(kraw[k]);
    float s   = sqrtf(fabsf(kap) + 1e-15f);
    // exp map + clip
    float tk    = tan_k(en, kap);
    float f     = tk / en;
    float pnorm = fabsf(tk);
    float maxn  = (kap < 0.f) ? (1.f - 1e-5f) / s : 1e15f;
    float scale = fminf(1.f, maxn / fmaxf(pnorm, EPSV));
    float pts[4];
#pragma unroll
    for (int j = 0; j < 4; ++j) pts[j] = e[j] * f * scale;
    float ptsn = fmaxf(pnorm * scale, EPSV);
    float x2   = ptsn * ptsn;
    // log map factor, scalar projection
    float lf = arctan_k(ptsn, kap) / ptsn;
    float dp = 0.f;
#pragma unroll
    for (int j = 0; j < 4; ++j) dp += pts[j] * w_hat[k * DD + lane + 32 * j];
    dp = wave_sum32(dp);
    float scalar = lf * dp;
    // interval memberships
    float t_iv = fmaxf(expf(lit[k]), 0.001f);
    float iv[NI], ivs = 0.f;
#pragma unroll
    for (int i = 0; i < NI; ++i) {
      float hwid = expf(lhw[k * NI + i]);
      float c = icen[k * NI + i];
      iv[i] = sigm((scalar - (c - hwid)) / t_iv) * sigm(((c + hwid) - scalar) / t_iv);
      ivs += iv[i];
    }
    float inv = 1.f / fmaxf(ivs, EPSV);
#pragma unroll
    for (int i = 0; i < NI; ++i) iv[i] *= inv;
    // prototype distances via Mobius addition
    float dist[NP];
#pragma unroll 1
    for (int p = 0; p < NP; ++p) {
      float y[4], xy = 0.f;
#pragma unroll
      for (int j = 0; j < 4; ++j) {
        y[j] = pm[(size_t)(k * NP + p) * DD + lane + 32 * j];
        xy += pts[j] * y[j];
      }
      xy = wave_sum32(xy);                 // dot(pts, pm)
      float y2 = pm_y2[k * NP + p];
      float mxy = -xy;                     // dot(-pts, pm)
      float c1 = 1.f - 2.f * kap * mxy - kap * y2;
      float c2 = 1.f + kap * x2;
      float den = 1.f - 2.f * kap * mxy + kap * kap * x2 * y2;
      den = (fabsf(den) < EPSV) ? EPSV : den;
      float nn = 0.f;
#pragma unroll
      for (int j = 0; j < 4; ++j) {
        float md = c1 * (-pts[j]) + c2 * y[j];
        nn += md * md;
      }
      nn = wave_sum32(nn);
      float man = sqrtf(nn) / fabsf(den);
      dist[p] = 2.f * arctan_k(man, kap);
    }
    // prototype softmax
    float tp = fmaxf(expf(lpt[k]), 0.001f);
    float pmax = -dist[0] / tp;
#pragma unroll
    for (int p = 1; p < NP; ++p) pmax = fmaxf(pmax, -dist[p] / tp);
    float pr[NP], psum = 0.f;
#pragma unroll
    for (int p = 0; p < NP; ++p) { pr[p] = expf(-dist[p] / tp - pmax); psum += pr[p]; }
#pragma unroll
    for (int p = 0; p < NP; ++p) pr[p] /= psum;
    // head: pred_k = quot @ head_w^T + head_b
    float acc[TT];
#pragma unroll
    for (int t = 0; t < TT; ++t) acc[t] = hb[k * TT + t];
#pragma unroll
    for (int i = 0; i < NI; ++i) {
#pragma unroll
      for (int p = 0; p < NP; ++p) {
        float q = iv[i] * pr[p];
#pragma unroll
        for (int t = 0; t < TT; ++t)
          acc[t] += q * hw[((size_t)k * TT + t) * (NI * NP) + i * NP + p];
      }
    }
#pragma unroll
    for (int t = 0; t < TT; ++t) pred[t] += sa[k] * acc[t];
  }

  if (lane == 0) {
#pragma unroll
    for (int t = 0; t < TT; ++t) out_pred[(size_t)node * TT + t] = pred[t];
#pragma unroll
    for (int k = 0; k < KK; ++k) out_sa[(size_t)node * KK + k] = sa[k];
  }
}

// ---------------- host orchestration ----------------

extern "C" void kernel_launch(void* const* d_in, const int* in_sizes, int n_in,
                              void* d_out, int out_size, void* d_ws, size_t ws_size,
                              hipStream_t stream) {
  (void)in_sizes; (void)n_in; (void)out_size; (void)ws_size;

  // ---- inputs (setup_inputs leaf order) ----
  const float* x    = (const float*)d_in[0];
  const int*   ei   = (const int*)d_in[1];
  const float* winp = (const float*)d_in[2];
  const float* binp = (const float*)d_in[3];
  const float* wl[NLAY];  const float* bl[NLAY];
  const float* wr[NLAY];  const float* br[NLAY];
  const float* lg[NLAY];  const float* lb[NLAY];
  for (int l = 0; l < NLAY; ++l) {
    wl[l] = (const float*)d_in[4 + 6 * l + 0];
    bl[l] = (const float*)d_in[4 + 6 * l + 1];
    wr[l] = (const float*)d_in[4 + 6 * l + 2];
    br[l] = (const float*)d_in[4 + 6 * l + 3];
    lg[l] = (const float*)d_in[4 + 6 * l + 4];
    lb[l] = (const float*)d_in[4 + 6 * l + 5];
  }
  const float* centroids = (const float*)d_in[28];
  const float* log_temp  = (const float*)d_in[29];
  const float* kraw      = (const float*)d_in[30];
  const float* icen      = (const float*)d_in[31];
  const float* lhw       = (const float*)d_in[32];
  const float* lit       = (const float*)d_in[33];
  const float* pw        = (const float*)d_in[34];
  const float* protos    = (const float*)d_in[35];
  const float* lpt       = (const float*)d_in[36];
  const float* hwd       = (const float*)d_in[37];
  const float* hbd       = (const float*)d_in[38];

  // ---- output layout: pred, soft_assign, embeds, protos_tan, kappas ----
  float* out_pred   = (float*)d_out;
  float* out_sa     = out_pred + (size_t)NN * TT;
  float* out_emb    = out_sa + (size_t)NN * KK;
  float* out_protos = out_emb + (size_t)NN * DD;
  float* out_kappas = out_protos + (size_t)KK * NP * DD;

  // ---- workspace carve (256B aligned) ----
  char* wsp = (char*)d_ws;
  size_t off = 0;
  auto carve = [&](size_t bytes) {
    char* p = wsp + off;
    off += (bytes + 255) & ~(size_t)255;
    return p;
  };
  float*    agg    = (float*)carve((size_t)NN * HH * 4);
  float*    hbuf   = (float*)carve((size_t)NN * HH * 4);
  float*    h2buf  = (float*)carve((size_t)NN * HH * 4);
  float*    deg    = (float*)carve((size_t)NN * 4);
  float*    pm     = (float*)carve((size_t)KK * NP * DD * 4);
  float*    pm_y2  = (float*)carve((size_t)KK * NP * 4);
  float*    w_hat  = (float*)carve((size_t)KK * DD * 4);
  _Float16* acat   = (_Float16*)carve((size_t)NN * 512 * 2);
  _Float16* wcat[NLAY];
  for (int l = 0; l < NLAY; ++l) {
    int oc = (l < NLAY - 1) ? HH : DD;
    wcat[l] = (_Float16*)carve((size_t)oc * 512 * 2);
  }

  // ---- graph setup ----
  zero_f32<<<(NN + 255) / 256, 256, 0, stream>>>(deg, NN);
  deg_kernel<<<(EE + 255) / 256, 256, 0, stream>>>(ei, deg);
  for (int l = 0; l < NLAY; ++l) {
    int oc = (l < NLAY - 1) ? HH : DD;
    wcat_prep<<<(oc * 512 + 255) / 256, 256, 0, stream>>>(wl[l], wr[l], wcat[l], oc);
  }
  input_proj<<<(NN * HH + 255) / 256, 256, 0, stream>>>(x, winp, binp, hbuf, acat);
  head_prep<<<1, 128, 0, stream>>>(kraw, pw, protos, w_hat, pm, pm_y2, out_kappas);
  copy_f32<<<(KK * NP * DD + 255) / 256, 256, 0, stream>>>(protos, out_protos, KK * NP * DD);

  for (int l = 0; l < NLAY; ++l) {
    int oc = (l < NLAY - 1) ? HH : DD;
    zero_f32<<<(NN * HH + 255) / 256, 256, 0, stream>>>(agg, NN * HH);
    scatter_add<<<(int)(((long long)EE * 64 + 255) / 256), 256, 0, stream>>>(hbuf, ei, agg);
    agg_norm<<<(NN * HH + 255) / 256, 256, 0, stream>>>(agg, deg, acat);
    int ntn = oc / 64;
    int blocks_m = ((NN / 16) + 7) / 8;               // 8 M-tiles per block
    gemm_wmma<<<blocks_m * ntn, 256, 0, stream>>>(acat, wcat[l], h2buf, oc);
    epilogue<<<NN, oc, 0, stream>>>(h2buf, bl[l], br[l], lg[l], lb[l],
                                    hbuf, acat, out_emb, oc, (l == NLAY - 1) ? 1 : 0);
  }

  head_kernel<<<(NN * 32 + 255) / 256, 256, 0, stream>>>(
      out_emb, centroids, log_temp, kraw, icen, lhw, lit,
      w_hat, pm, pm_y2, lpt, hwd, hbd, out_pred, out_sa);
}
